// RecurrentStateSpaceModel_84207128805695
// MI455X (gfx1250) — matrix operand order repaired and checked by reading, not compiled, not run on hardware
//
#include <hip/hip_runtime.h>
#include <cstddef>
#include <cstdint>

// ---------------- model dims ----------------
#define NB   32      // batch
#define TT   50      // timesteps
#define ASZ  6       // action size
#define SSZ  256     // state size
#define LSZ  64      // latent size
#define ESZ  1024    // embedding size
#define HIDZ 256     // hidden size
#define NT   (NB*TT) // 1600 frames

// ---------------- WMMA types ----------------
typedef __attribute__((ext_vector_type(16))) _Float16 v16h;
typedef __attribute__((ext_vector_type(8)))  _Float16 v8h_t;
typedef __attribute__((ext_vector_type(8)))  float    v8f;

union FragU { v16h v; v8h_t h[2]; _Float16 e[16]; };

__device__ inline v8f wmma_f16(v16h a, v16h b, v8f c) {
    return __builtin_amdgcn_wmma_f32_16x16x32_f16(
        false, a, false, b, (short)0, c, false, false);
}

// A fragment: lane holds row (rowb + lane&15); chunks at K+(lane>>4)*8 and +16
__device__ inline v16h load_a_frag(const _Float16* A, int lda,
                                   int rowb, int kb, int lane) {
    int m    = rowb + (lane & 15);
    int koff = (lane >> 4) * 8;
    const _Float16* p = A + (size_t)m * lda + kb + koff;
    FragU u;
    u.h[0] = *(const v8h_t*)(p);
    u.h[1] = *(const v8h_t*)(p + 16);
    return u.v;
}

// B fragment for x @ W^T, W[N,K] row-major: lane holds W-row (colb + lane&15),
// chunks at kb + (lane>>4)*8 and +16
__device__ inline v16h load_b_frag(const _Float16* W, int ldw,
                                   int colb, int kb, int lane) {
    int n    = colb + (lane & 15);
    int koff = (lane >> 4) * 8;
    const _Float16* p = W + (size_t)n * ldw + kb + koff;
    FragU u;
    u.h[0] = *(const v8h_t*)(p);
    u.h[1] = *(const v8h_t*)(p + 16);
    return u.v;
}

__device__ inline float sigmoidf_(float x) { return 1.0f / (1.0f + __expf(-x)); }
__device__ inline float softplusf_(float x) { return (x > 20.0f) ? x : log1pf(__expf(x)); }

// ---------------- async global -> LDS copy (CDNA5 ASYNCcnt path) ----------------
#if defined(__has_builtin)
#if __has_builtin(__builtin_amdgcn_global_load_async_to_lds_b128) && \
    __has_builtin(__builtin_amdgcn_s_wait_asynccnt)
#define USE_ASYNC_LDS 1
#endif
#endif

typedef int v4i_gcc_t __attribute__((__vector_size__(16)));

__device__ inline void async_cp16(const _Float16* g, _Float16* l) {
#ifdef USE_ASYNC_LDS
    __builtin_amdgcn_global_load_async_to_lds_b128(
        (__attribute__((address_space(1))) v4i_gcc_t*)(g),
        (__attribute__((address_space(3))) v4i_gcc_t*)(l), 0, 0);
#else
    *(v8h_t*)l = *(const v8h_t*)g;   // fallback: load + ds_store (DScnt, drained by barrier)
#endif
}

__device__ inline void async_wait_all() {
#ifdef USE_ASYNC_LDS
    __builtin_amdgcn_s_wait_asynccnt(0);
#endif
}

// ---------------- f32 -> f16 convert with optional K padding ----------------
__global__ void cvt_pad_kernel(const float* __restrict__ src, _Float16* __restrict__ dst,
                               long rows, int Kin, int Kpad) {
    long i = (long)blockIdx.x * blockDim.x + threadIdx.x;
    long tot = rows * (long)Kpad;
    if (i >= tot) return;
    long r = i / Kpad;
    int  k = (int)(i - r * Kpad);
    dst[i] = (k < Kin) ? (_Float16)src[r * (long)Kin + k] : (_Float16)0.0f;
}

// ---------------- LDS-staged, double-buffered WMMA GEMM ----------------
// D = act(A[M,K] @ W[N,K]^T + bias). 256 thr = 8 waves, block tile 64x64,
// wave = 32x16 (2 accs sharing one B frag). K staged in 64-wide slices via
// async global->LDS copies; LDS rows padded to 72 halfs (bank-conflict-free).
#define KS 64
#define LSTR 72

__global__ __launch_bounds__(256)
void gemm_wmma_lds_kernel(const _Float16* __restrict__ A, int lda,
                          const _Float16* __restrict__ W, int ldw,
                          const float* __restrict__ bias,
                          _Float16* __restrict__ out16, float* __restrict__ out32,
                          int ldo, int K, int act) {
    __shared__ __align__(16) _Float16 As[2][64 * LSTR];
    __shared__ __align__(16) _Float16 Bs[2][64 * LSTR];
    int tid = threadIdx.x;
    int lane = tid & 31, wave = tid >> 5;
    int mb = blockIdx.x * 64, nb = blockIdx.y * 64;

    // staging: each thread moves 4 x b128 (A: rows r, r+32; B: rows r, r+32)
    int sr = tid >> 3, sc = (tid & 7) * 8;
    auto stage = [&](int buf, int kb) {
        async_cp16(A + (size_t)(mb + sr)      * lda + kb + sc, &As[buf][sr * LSTR + sc]);
        async_cp16(A + (size_t)(mb + sr + 32) * lda + kb + sc, &As[buf][(sr + 32) * LSTR + sc]);
        async_cp16(W + (size_t)(nb + sr)      * ldw + kb + sc, &Bs[buf][sr * LSTR + sc]);
        async_cp16(W + (size_t)(nb + sr + 32) * ldw + kb + sc, &Bs[buf][(sr + 32) * LSTR + sc]);
    };

    int wm = wave & 1, wn = wave >> 1;   // 2 (M) x 4 (N) wave grid
    int rowb = wm * 32;                  // two 16-row tiles: rowb, rowb+16
    int colb = wn * 16;

    v8f acc0 = {}, acc1 = {};
    stage(0, 0);
    int nstages = K / KS;
    for (int s = 0; s < nstages; ++s) {
        async_wait_all();
        __syncthreads();     // staged slice visible; prev compute finished everywhere
        if (s + 1 < nstages) stage((s + 1) & 1, (s + 1) * KS);
        const _Float16* Ab = &As[s & 1][0];
        const _Float16* Bb = &Bs[s & 1][0];
#pragma unroll
        for (int kb = 0; kb < KS; kb += 32) {
            v16h b  = load_b_frag(Bb, LSTR, colb, kb, lane);
            v16h a0 = load_a_frag(Ab, LSTR, rowb, kb, lane);
            v16h a1 = load_a_frag(Ab, LSTR, rowb + 16, kb, lane);
            acc0 = wmma_f16(a0, b, acc0);
            acc1 = wmma_f16(a1, b, acc1);
        }
    }
    int col = nb + colb + (lane & 15);
    float bv = bias ? bias[col] : 0.0f;
#pragma unroll
    for (int r = 0; r < 8; ++r) {
        int row = mb + rowb + r + 8 * (lane >> 4);
        float v0 = acc0[r] + bv;
        float v1 = acc1[r] + bv;
        if (act == 1) { v0 = fmaxf(v0, 0.0f); v1 = fmaxf(v1, 0.0f); }
        if (out16) {
            out16[(size_t)row * ldo + col]        = (_Float16)v0;
            out16[(size_t)(row + 16) * ldo + col] = (_Float16)v1;
        } else {
            out32[(size_t)row * ldo + col]        = v0;
            out32[(size_t)(row + 16) * ldo + col] = v1;
        }
    }
}

// ---------------- implicit-im2col conv (4x4, stride 2, VALID) + bias + relu --------
// Each wave: 16(M) x 32(N), 2 accs reuse the gathered A fragment (gathers dominate).
// Block: 4 waves -> tile 64(M) x 32(N). in/out NCHW f16, weights [OC, Kpad] f16.
__global__ __launch_bounds__(128)
void conv_wmma_kernel(const _Float16* __restrict__ in, const _Float16* __restrict__ wgt,
                      const float* __restrict__ bias, _Float16* __restrict__ out,
                      int IC, int IH, int IW, int OC, int OH, int OW,
                      int Kreal, int Kpad) {
    int lane = threadIdx.x & 31;
    int wave = threadIdx.x >> 5;
    int rowb  = blockIdx.x * 64 + wave * 16;
    int colb0 = blockIdx.y * 32;

    int opix = OH * OW;
    int m  = rowb + (lane & 15);
    int f  = m / opix, pr = m - f * opix;
    int oy = pr / OW,  ox = pr - oy * OW;
    const _Float16* inb = in + (((size_t)f * IC) * IH + oy * 2) * (size_t)IW + ox * 2;
    int chs = IH * IW;

    v8f acc0 = {}, acc1 = {};
    for (int kb = 0; kb < Kpad; kb += 32) {
        FragU a;
        int k0 = kb + (lane >> 4) * 8;
#pragma unroll
        for (int i = 0; i < 8; ++i) {
            int k = k0 + i;
            _Float16 v = (_Float16)0.0f;
            if (k < Kreal) {
                int ic = k >> 4, rem = k & 15;
                v = inb[(size_t)ic * chs + (rem >> 2) * IW + (rem & 3)];
            }
            a.e[i] = v;
            int k2 = k + 16;
            _Float16 v2 = (_Float16)0.0f;
            if (k2 < Kreal) {
                int ic = k2 >> 4, rem = k2 & 15;
                v2 = inb[(size_t)ic * chs + (rem >> 2) * IW + (rem & 3)];
            }
            a.e[8 + i] = v2;
        }
        v16h b0 = load_b_frag(wgt, Kpad, colb0, kb, lane);
        v16h b1 = load_b_frag(wgt, Kpad, colb0 + 16, kb, lane);
        acc0 = wmma_f16(a.v, b0, acc0);
        acc1 = wmma_f16(a.v, b1, acc1);
    }
    int oc0 = colb0 + (lane & 15);
    float bv0 = bias[oc0], bv1 = bias[oc0 + 16];
#pragma unroll
    for (int r = 0; r < 8; ++r) {
        int row = rowb + r + 8 * (lane >> 4);
        int f2 = row / opix, pr2 = row - f2 * opix;
        int oy2 = pr2 / OW, ox2 = pr2 - oy2 * OW;
        size_t base = (((size_t)f2 * OC) * OH + oy2) * (size_t)OW + ox2;
        out[base + (size_t)oc0 * opix]        = (_Float16)fmaxf(acc0[r] + bv0, 0.0f);
        out[base + (size_t)(oc0 + 16) * opix] = (_Float16)fmaxf(acc1[r] + bv1, 0.0f);
    }
}

// ---------------- workgroup cooperative GEMM for the scan (M fixed = 32) ----------------
__device__ void wg_gemm32(const _Float16* A, int lda,
                          const _Float16* __restrict__ W, int ldw,
                          const float* __restrict__ bias, int N, int K,
                          _Float16* out16, float* out32, int ldo, int act, int tid) {
    int lane = tid & 31;
    int wave = tid >> 5;                    // 0..7
    int ntiles = 2 * (N >> 4);
    for (int tl = wave; tl < ntiles; tl += 8) {
        int rowb = (tl & 1) * 16;
        int colb = (tl >> 1) * 16;
        v8f acc = {};
        for (int kb = 0; kb < K; kb += 32) {
            v16h a = load_a_frag(A, lda, rowb, kb, lane);
            v16h b = load_b_frag(W, ldw, colb, kb, lane);
            acc = wmma_f16(a, b, acc);
        }
        int col = colb + (lane & 15);
        float bv = bias ? bias[col] : 0.0f;
#pragma unroll
        for (int r = 0; r < 8; ++r) {
            int row = rowb + r + 8 * (lane >> 4);
            float v = acc[r] + bv;
            if (act == 1) v = fmaxf(v, 0.0f);
            if (out16) out16[row * ldo + col] = (_Float16)v;
            else       out32[(size_t)row * ldo + col] = v;
        }
    }
}

// ---------------- persistent sequential scan: init state + 49 GRU steps ----------------
__global__ __launch_bounds__(256)
void scan_kernel(const _Float16* __restrict__ en,
                 const float* __restrict__ actions,
                 const float* __restrict__ noise_init,
                 const float* __restrict__ noise_prior,
                 const _Float16* w_init1, const float* b_init1,
                 const _Float16* w_init2, const float* b_init2,
                 const _Float16* w_prior1, const float* b_prior1,
                 const _Float16* w_prior2, const float* b_prior2,
                 const _Float16* w_cla1, const float* b_cla1,
                 const _Float16* w_cla2, const float* b_cla2,
                 const _Float16* w_ih, const float* b_ih,
                 const _Float16* w_hh, const float* b_hh,
                 float* ws_io, float* ws_gi, float* ws_gh,
                 float* ws_o, float* ws_h,
                 float* out_states, float* out_pm, float* out_ps, float* out_pz) {
    __shared__ __align__(16) _Float16 s_h[NB * SSZ];
    __shared__ __align__(16) _Float16 s_bufA[NB * HIDZ];
    __shared__ __align__(16) _Float16 s_bufB[NB * SSZ];
    __shared__ __align__(16) _Float16 s_z[NB * 96];
    int tid = threadIdx.x;

    // warm L2 for the serial section's hot weights (global_prefetch_b8)
    for (int i = tid; i < (3 * SSZ * SSZ) / 64; i += 256) {
        __builtin_prefetch(w_ih + (size_t)i * 64, 0, 1);
        __builtin_prefetch(w_hh + (size_t)i * 64, 0, 1);
    }
    for (int i = tid; i < NB * 96; i += 256) s_z[i] = (_Float16)0.0f;
    __syncthreads();

    // ---- initial state: s0 = sm + noise_init * (softplus(sd)+0.1), from en[:,0] ----
    wg_gemm32(en, TT * ESZ, w_init1, ESZ, b_init1, HIDZ, ESZ, s_bufA, nullptr, HIDZ, 1, tid);
    __syncthreads();
    wg_gemm32(s_bufA, HIDZ, w_init2, HIDZ, b_init2, 2 * SSZ, HIDZ, nullptr, ws_io, 2 * SSZ, 0, tid);
    __syncthreads();
    for (int i = tid; i < NB * SSZ; i += 256) {
        int r = i >> 8, j = i & 255;
        float sm = ws_io[r * 512 + j];
        float sd = softplusf_(ws_io[r * 512 + 256 + j]) + 0.1f;
        float s0 = sm + noise_init[i] * sd;
        ws_h[i] = s0;
        s_h[i] = (_Float16)s0;
        out_states[((size_t)r * TT + 0) * SSZ + j] = s0;
    }
    __syncthreads();

    for (int t = 0; t < TT; ++t) {
        if (t > 0) {
            wg_gemm32(s_z, 96, w_cla1, 96, b_cla1, HIDZ, 96, s_bufA, nullptr, HIDZ, 1, tid);
            __syncthreads();
            wg_gemm32(s_bufA, HIDZ, w_cla2, HIDZ, b_cla2, SSZ, HIDZ, s_bufB, nullptr, SSZ, 0, tid);
            __syncthreads();
            wg_gemm32(s_bufB, SSZ, w_ih, SSZ, b_ih, 3 * SSZ, SSZ, nullptr, ws_gi, 3 * SSZ, 0, tid);
            wg_gemm32(s_h,   SSZ, w_hh, SSZ, b_hh, 3 * SSZ, SSZ, nullptr, ws_gh, 3 * SSZ, 0, tid);
            __syncthreads();
            for (int i = tid; i < NB * SSZ; i += 256) {
                int r = i >> 8, j = i & 255;
                float ir = ws_gi[r * 768 + j], iz = ws_gi[r * 768 + 256 + j], ig = ws_gi[r * 768 + 512 + j];
                float hr = ws_gh[r * 768 + j], hz = ws_gh[r * 768 + 256 + j], hg = ws_gh[r * 768 + 512 + j];
                float rr = sigmoidf_(ir + hr);
                float zz = sigmoidf_(iz + hz);
                float g  = tanhf(ig + rr * hg);
                float h2 = (1.0f - zz) * g + zz * ws_h[i];
                ws_h[i] = h2;
                s_h[i] = (_Float16)h2;
                out_states[((size_t)r * TT + t) * SSZ + j] = h2;
            }
            __syncthreads();
        }
        wg_gemm32(s_h, SSZ, w_prior1, SSZ, b_prior1, HIDZ, SSZ, s_bufA, nullptr, HIDZ, 1, tid);
        __syncthreads();
        wg_gemm32(s_bufA, HIDZ, w_prior2, HIDZ, b_prior2, 2 * LSZ, HIDZ, nullptr, ws_o, 2 * LSZ, 0, tid);
        __syncthreads();
        for (int i = tid; i < NB * LSZ; i += 256) {
            int r = i >> 6, j = i & 63;
            float pm = ws_o[r * 128 + j];
            float ps = softplusf_(ws_o[r * 128 + 64 + j]) + 0.1f;
            size_t oi = ((size_t)r * TT + t) * LSZ + j;
            float z = pm + noise_prior[oi] * ps;
            out_pm[oi] = pm; out_ps[oi] = ps; out_pz[oi] = z;
            s_z[r * 96 + j] = (_Float16)z;
        }
        if (t < TT - 1) {
            for (int i = tid; i < NB * ASZ; i += 256) {
                int r = i / ASZ, k = i - r * ASZ;
                s_z[r * 96 + 64 + k] = (_Float16)actions[((size_t)r * (TT - 1) + t) * ASZ + k];
            }
        }
        __syncthreads();
    }
}

// ---------------- posterior input concat [states(f32->f16) | en] ----------------
__global__ void concat_post_kernel(const float* __restrict__ states,
                                   const _Float16* __restrict__ en,
                                   _Float16* __restrict__ out) {
    long i = (long)blockIdx.x * blockDim.x + threadIdx.x;
    if (i >= (long)NT * (SSZ + ESZ)) return;
    long r = i / (SSZ + ESZ);
    int  c = (int)(i - r * (SSZ + ESZ));
    out[i] = (c < SSZ) ? (_Float16)states[r * SSZ + c] : en[r * (long)ESZ + (c - SSZ)];
}

// ---------------- posterior elementwise: qm, qs, q_samples ----------------
__global__ void post_elem_kernel(const float* __restrict__ o,
                                 const float* __restrict__ noise_post,
                                 float* __restrict__ qm, float* __restrict__ qs,
                                 float* __restrict__ qsamp) {
    int i = blockIdx.x * blockDim.x + threadIdx.x;
    if (i >= NT * LSZ) return;
    int r = i >> 6, j = i & 63;
    float m = o[r * 128 + j];
    float s = softplusf_(o[r * 128 + 64 + j]) + 0.1f;
    qm[i] = m; qs[i] = s; qsamp[i] = m + noise_post[i] * s;
}

// ======================= launcher =======================
extern "C" void kernel_launch(void* const* d_in, const int* in_sizes, int n_in,
                              void* d_out, int out_size, void* d_ws, size_t ws_size,
                              hipStream_t stream) {
    (void)in_sizes; (void)n_in; (void)out_size; (void)ws_size;
    const float* obs        = (const float*)d_in[0];
    const float* actions    = (const float*)d_in[1];
    const float* noise_init = (const float*)d_in[2];
    const float* noise_pri  = (const float*)d_in[3];
    const float* noise_post = (const float*)d_in[4];
    const float* enc_w1 = (const float*)d_in[5];  const float* enc_b1 = (const float*)d_in[6];
    const float* enc_w2 = (const float*)d_in[7];  const float* enc_b2 = (const float*)d_in[8];
    const float* enc_w3 = (const float*)d_in[9];  const float* enc_b3 = (const float*)d_in[10];
    const float* enc_w4 = (const float*)d_in[11]; const float* enc_b4 = (const float*)d_in[12];
    const float* fc_w   = (const float*)d_in[13]; const float* fc_b   = (const float*)d_in[14];
    const float* in_w1  = (const float*)d_in[15]; const float* in_b1  = (const float*)d_in[16];
    const float* in_w2  = (const float*)d_in[17]; const float* in_b2  = (const float*)d_in[18];
    const float* pr_w1  = (const float*)d_in[19]; const float* pr_b1  = (const float*)d_in[20];
    const float* pr_w2  = (const float*)d_in[21]; const float* pr_b2  = (const float*)d_in[22];
    const float* po_w1  = (const float*)d_in[23]; const float* po_b1  = (const float*)d_in[24];
    const float* po_w2  = (const float*)d_in[25]; const float* po_b2  = (const float*)d_in[26];
    const float* cl_w1  = (const float*)d_in[27]; const float* cl_b1  = (const float*)d_in[28];
    const float* cl_w2  = (const float*)d_in[29]; const float* cl_b2  = (const float*)d_in[30];
    const float* g_wih  = (const float*)d_in[31]; const float* g_whh  = (const float*)d_in[32];
    const float* g_bih  = (const float*)d_in[33]; const float* g_bhh  = (const float*)d_in[34];

    float* out_states = (float*)d_out;
    float* out_pm     = out_states + (size_t)NT * SSZ;
    float* out_ps     = out_pm + (size_t)NT * LSZ;
    float* out_pz     = out_ps + (size_t)NT * LSZ;
    float* out_qm     = out_pz + (size_t)NT * LSZ;
    float* out_qs     = out_qm + (size_t)NT * LSZ;
    float* out_qsamp  = out_qs + (size_t)NT * LSZ;

    char* ws = (char*)d_ws;
    size_t off = 0;
    auto allocH = [&](size_t n) { _Float16* p = (_Float16*)(ws + off);
                                  off = (off + n * 2 + 255) & ~(size_t)255; return p; };
    auto allocF = [&](size_t n) { float* p = (float*)(ws + off);
                                  off = (off + n * 4 + 255) & ~(size_t)255; return p; };

    _Float16* obs16 = allocH((size_t)NT * 3 * 64 * 64);
    _Float16* act1  = allocH((size_t)NT * 32 * 31 * 31);
    _Float16* act2  = allocH((size_t)NT * 64 * 14 * 14);
    _Float16* act3  = allocH((size_t)NT * 128 * 6 * 6);
    _Float16* act4  = allocH((size_t)NT * 256 * 2 * 2);
    _Float16* en16  = allocH((size_t)NT * ESZ);
    _Float16* wc1 = allocH(32 * 64);
    _Float16* wc2 = allocH(64 * 512);
    _Float16* wc3 = allocH(128 * 1024);
    _Float16* wc4 = allocH(256 * 2048);
    _Float16* wfc = allocH(1024 * 1024);
    _Float16* wi1 = allocH((size_t)HIDZ * ESZ);
    _Float16* wi2 = allocH((size_t)2 * SSZ * HIDZ);
    _Float16* wp1 = allocH((size_t)HIDZ * SSZ);
    _Float16* wp2 = allocH((size_t)2 * LSZ * HIDZ);
    _Float16* wq1 = allocH((size_t)HIDZ * (SSZ + ESZ));
    _Float16* wq2 = allocH((size_t)2 * LSZ * HIDZ);
    _Float16* wl1 = allocH((size_t)HIDZ * 96);
    _Float16* wl2 = allocH((size_t)SSZ * HIDZ);
    _Float16* wih = allocH((size_t)3 * SSZ * SSZ);
    _Float16* whh = allocH((size_t)3 * SSZ * SSZ);
    float* s_io = allocF(NB * 512);
    float* s_gi = allocF(NB * 768);
    float* s_gh = allocF(NB * 768);
    float* s_o  = allocF(NB * 128);
    float* s_h  = allocF(NB * 256);
    _Float16* post_in = allocH((size_t)NT * (SSZ + ESZ));
    _Float16* post_h  = allocH((size_t)NT * HIDZ);
    float*    post_o  = allocF((size_t)NT * 2 * LSZ);

    auto cvt = [&](const float* src, _Float16* dst, long rows, int kin, int kpad) {
        long tot = rows * (long)kpad;
        int blocks = (int)((tot + 255) / 256);
        cvt_pad_kernel<<<blocks, 256, 0, stream>>>(src, dst, rows, kin, kpad);
    };

    cvt(obs,    obs16, 1, NT * 3 * 64 * 64, NT * 3 * 64 * 64);
    cvt(enc_w1, wc1, 32, 48, 64);
    cvt(enc_w2, wc2, 1, 64 * 512, 64 * 512);
    cvt(enc_w3, wc3, 1, 128 * 1024, 128 * 1024);
    cvt(enc_w4, wc4, 1, 256 * 2048, 256 * 2048);
    cvt(fc_w,   wfc, 1, 1024 * 1024, 1024 * 1024);
    cvt(in_w1,  wi1, 1, HIDZ * ESZ, HIDZ * ESZ);
    cvt(in_w2,  wi2, 1, 2 * SSZ * HIDZ, 2 * SSZ * HIDZ);
    cvt(pr_w1,  wp1, 1, HIDZ * SSZ, HIDZ * SSZ);
    cvt(pr_w2,  wp2, 1, 2 * LSZ * HIDZ, 2 * LSZ * HIDZ);
    cvt(po_w1,  wq1, 1, HIDZ * (SSZ + ESZ), HIDZ * (SSZ + ESZ));
    cvt(po_w2,  wq2, 1, 2 * LSZ * HIDZ, 2 * LSZ * HIDZ);
    cvt(cl_w1,  wl1, HIDZ, LSZ + ASZ, 96);
    cvt(cl_w2,  wl2, 1, SSZ * HIDZ, SSZ * HIDZ);
    cvt(g_wih,  wih, 1, 3 * SSZ * SSZ, 3 * SSZ * SSZ);
    cvt(g_whh,  whh, 1, 3 * SSZ * SSZ, 3 * SSZ * SSZ);

    // ---- encoder: 4 implicit-GEMM convs ----
    {   int M = NT * 31 * 31;   // conv1: 3x64x64 -> 32x31x31
        conv_wmma_kernel<<<dim3(M / 64, 32 / 32), 128, 0, stream>>>(
            obs16, wc1, enc_b1, act1, 3, 64, 64, 32, 31, 31, 48, 64);
    }
    {   int M = NT * 14 * 14;   // conv2: 32x31x31 -> 64x14x14
        conv_wmma_kernel<<<dim3(M / 64, 64 / 32), 128, 0, stream>>>(
            act1, wc2, enc_b2, act2, 32, 31, 31, 64, 14, 14, 512, 512);
    }
    {   int M = NT * 6 * 6;     // conv3: 64x14x14 -> 128x6x6
        conv_wmma_kernel<<<dim3(M / 64, 128 / 32), 128, 0, stream>>>(
            act2, wc3, enc_b3, act3, 64, 14, 14, 128, 6, 6, 1024, 1024);
    }
    {   int M = NT * 2 * 2;     // conv4: 128x6x6 -> 256x2x2
        conv_wmma_kernel<<<dim3(M / 64, 256 / 32), 128, 0, stream>>>(
            act3, wc4, enc_b4, act4, 128, 6, 6, 256, 2, 2, 2048, 2048);
    }
    // enc_fc: [1600,1024] @ [1024,1024]^T -> en (f16)
    gemm_wmma_lds_kernel<<<dim3(NT / 64, ESZ / 64), 256, 0, stream>>>(
        act4, 1024, wfc, 1024, fc_b, en16, nullptr, ESZ, 1024, 0);

    // ---- persistent sequential scan ----
    scan_kernel<<<1, 256, 0, stream>>>(
        en16, actions, noise_init, noise_pri,
        wi1, in_b1, wi2, in_b2, wp1, pr_b1, wp2, pr_b2,
        wl1, cl_b1, wl2, cl_b2, wih, g_bih, whh, g_bhh,
        s_io, s_gi, s_gh, s_o, s_h,
        out_states, out_pm, out_ps, out_pz);

    // ---- posterior ----
    {
        long tot = (long)NT * (SSZ + ESZ);
        concat_post_kernel<<<(int)((tot + 255) / 256), 256, 0, stream>>>(
            out_states, en16, post_in);
    }
    gemm_wmma_lds_kernel<<<dim3(NT / 64, HIDZ / 64), 256, 0, stream>>>(
        post_in, SSZ + ESZ, wq1, SSZ + ESZ, po_b1, post_h, nullptr, HIDZ, SSZ + ESZ, 1);
    gemm_wmma_lds_kernel<<<dim3(NT / 64, (2 * LSZ) / 64), 256, 0, stream>>>(
        post_h, HIDZ, wq2, HIDZ, po_b2, nullptr, post_o, 2 * LSZ, HIDZ, 0);
    post_elem_kernel<<<(NT * LSZ + 255) / 256, 256, 0, stream>>>(
        post_o, noise_post, out_qm, out_qs, out_qsamp);
}
